// SpikingLinear_su_27169963115022
// MI455X (gfx1250) — matrix operand order repaired and checked by reading, not compiled, not run on hardware
//
#include <hip/hip_runtime.h>
#include <stdint.h>

typedef __attribute__((ext_vector_type(16))) _Float16 v16h;
typedef __attribute__((ext_vector_type(8)))  _Float16 v8h;
typedef __attribute__((ext_vector_type(4)))  _Float16 v4h;
typedef __attribute__((ext_vector_type(8)))  float    v8f;
typedef __attribute__((ext_vector_type(4)))  float    v4f;

#define T_    128
#define B_    256
#define D1_   1024
#define D2_   1024
#define MTOT  (T_ * B_)            // 32768 rows of A / C
#define TBD   ((size_t)T_ * B_ * D2_)

// ---------------------------------------------------------------------------
// Kernel 1: W [D2][D1] f32 -> f16 (row-major preserved; rows are contiguous K,
// which is exactly the WMMA B-fragment per-lane layout).
// ---------------------------------------------------------------------------
__global__ void w_to_f16(const float* __restrict__ W, _Float16* __restrict__ Wh) {
    size_t gid = (size_t)blockIdx.x * blockDim.x + threadIdx.x;
    v4f w = *(const v4f*)(W + gid * 4);
    v4h h;
    h[0] = (_Float16)w[0];
    h[1] = (_Float16)w[1];
    h[2] = (_Float16)w[2];
    h[3] = (_Float16)w[3];
    *(v4h*)(Wh + gid * 4) = h;
}

// ---------------------------------------------------------------------------
// Kernel 2: cur[M][D2] = X[M][D1] (f32, binary) @ Wh^T  via v_wmma_f32_16x16x32_f16
//   block tile: 128(M) x 128(N), 8 waves of 32, wave tile 32(M) x 64(N)
//   K loop: 32 steps of 32; A tile (128x32 f32) staged to LDS with
//   global_load_async_to_lds_b128, double buffered, f32->f16 cvt on frag read.
//   B fragments loaded directly from Wh (f16, L2-resident, 2 MB).
// ---------------------------------------------------------------------------
__launch_bounds__(256)
__global__ void snn_gemm(const float* __restrict__ X,
                         const _Float16* __restrict__ Wh,
                         float* __restrict__ Cur) {
    __shared__ float As[2][128 * 32];   // 2 x 16 KB

    const int tid   = threadIdx.x;
    const int lane  = tid & 31;
    const int wave  = tid >> 5;
    const int waveM = wave & 3;         // 0..3  -> 32-row strip
    const int waveN = wave >> 2;        // 0..1  -> 64-col strip

    const int nt = blockIdx.x & 7;      // fastest: N tiles share the A strip
    const int mt = blockIdx.x >> 3;
    const int mBase = mt * 128;
    const int nBase = nt * 128;

    const int rowSel = lane >> 4;       // 0 | 1 (half-wave)
    const int lane16 = lane & 15;
    const int aC0 = rowSel * 8;         // A: K chunks {c0..c0+7, c0+16..c0+23}
    const int bC0 = rowSel * 8;         // B: K chunks {c0..c0+7, c0+16..c0+23}

    // issue one double-buffer stage of A: 128 rows x 32 K of f32 = 16 KB
    auto stage = [&](int k0, int buf) {
        const float* src = X + (size_t)mBase * D1_ + k0;
        uint32_t ldsBase = (uint32_t)(uintptr_t)(&As[buf][0]);
#pragma unroll
        for (int j = 0; j < 4; ++j) {
            int chunk = tid + j * 256;              // 1024 chunks of 16B
            int row = chunk >> 3;
            int seg = chunk & 7;
            uint64_t ga = (uint64_t)(uintptr_t)(src + (size_t)row * D1_ + seg * 4);
            uint32_t la = ldsBase + (uint32_t)(row * 128 + seg * 16);
            asm volatile("global_load_async_to_lds_b128 %0, %1, off"
                         :: "v"(la), "v"(ga) : "memory");
        }
    };

    v8f acc[2][4] = {};

    stage(0, 0);

    for (int it = 0; it < 32; ++it) {
        const int buf = it & 1;
        const int k0  = it * 32;

        asm volatile("s_wait_asynccnt 0" ::: "memory");
        __syncthreads();                       // staged A visible to all waves
        if (it + 1 < 32) stage(k0 + 32, buf ^ 1);

        // A fragments: f32 from LDS, convert to f16 in the WMMA lane layout
        v16h a[2];
#pragma unroll
        for (int mf = 0; mf < 2; ++mf) {
            const int row = waveM * 32 + mf * 16 + lane16;
            const float* ap = &As[buf][row * 32 + aC0];
#pragma unroll
            for (int e = 0; e < 8; ++e) a[mf][e]     = (_Float16)ap[e];
#pragma unroll
            for (int e = 0; e < 8; ++e) a[mf][8 + e] = (_Float16)ap[16 + e];
        }

        // B fragments: two 16-byte chunks per lane from Wh row n
        v16h b[4];
#pragma unroll
        for (int nf = 0; nf < 4; ++nf) {
            const int n = nBase + waveN * 64 + nf * 16 + lane16;
            const _Float16* bp = Wh + (size_t)n * D1_ + k0 + bC0;
            v8h lo = *(const v8h*)(bp);
            v8h hi = *(const v8h*)(bp + 16);
#pragma unroll
            for (int e = 0; e < 8; ++e) { b[nf][e] = lo[e]; b[nf][8 + e] = hi[e]; }
        }

#pragma unroll
        for (int mf = 0; mf < 2; ++mf)
#pragma unroll
            for (int nf = 0; nf < 4; ++nf)
                acc[mf][nf] = __builtin_amdgcn_wmma_f32_16x16x32_f16(
                    false, a[mf], false, b[nf],
                    (short)0, acc[mf][nf], false, false);
    }

    // epilogue: C VGPR r -> M = r + 8*rowSel, N = lane16
#pragma unroll
    for (int mf = 0; mf < 2; ++mf)
#pragma unroll
        for (int nf = 0; nf < 4; ++nf) {
            const int col = nBase + waveN * 64 + nf * 16 + lane16;
            const int row = mBase + waveM * 32 + mf * 16 + rowSel * 8;
            float* cp = Cur + (size_t)row * D2_ + col;
#pragma unroll
            for (int r = 0; r < 8; ++r)
                cp[(size_t)r * D2_] = acc[mf][nf][r];
        }
}

// ---------------------------------------------------------------------------
// Kernel 3: LIF scan over T. One thread per (b, o). Reads cur (pre-written into
// the currents output slice) and overwrites it in place with syn.
// ---------------------------------------------------------------------------
__global__ void snn_scan(float* __restrict__ spk,
                         float* __restrict__ vmem,
                         float* __restrict__ cur) {
    const float ALPHA = 0.8187307530779818f;   // exp(-1/5)
    const float BETA  = 0.9048374180359595f;   // exp(-1/10)
    const size_t stride = (size_t)B_ * D2_;

    size_t off = (size_t)blockIdx.x * blockDim.x + threadIdx.x;
    float syn = 0.0f, mem = 0.0f;
#pragma unroll 4
    for (int t = 0; t < T_; ++t, off += stride) {
        float c = cur[off];
        float reset = (mem > 1.0f) ? 1.0f : 0.0f;
        syn = ALPHA * syn + c;
        mem = (1.0f - reset) * (BETA * mem + syn);
        float s = (mem > 1.0f) ? 1.0f : 0.0f;
        cur[off]  = syn;
        vmem[off] = mem;
        spk[off]  = s;
    }
}

// ---------------------------------------------------------------------------
extern "C" void kernel_launch(void* const* d_in, const int* in_sizes, int n_in,
                              void* d_out, int out_size, void* d_ws, size_t ws_size,
                              hipStream_t stream) {
    const float* x = (const float*)d_in[0];   // [T,B,D1] = [32768,1024] row-major
    const float* W = (const float*)d_in[1];   // [D2,D1]  row-major

    float* out  = (float*)d_out;
    float* spk  = out;                        // [T,B,D2]
    float* vmem = out + TBD;                  // [T,B,D2]
    float* cur  = out + 2 * TBD;              // [T,B,D2]  (cur, then syn in place)

    _Float16* Wh = (_Float16*)d_ws;           // 2 MB scratch

    // 1) W -> f16 (1M elems, 4 per thread)
    w_to_f16<<<(D2_ * D1_) / (256 * 4), 256, 0, stream>>>(W, Wh);

    // 2) GEMM: (32768/128) x (1024/128) = 256 x 8 = 2048 blocks
    snn_gemm<<<(MTOT / 128) * (D2_ / 128), 256, 0, stream>>>(x, Wh, cur);

    // 3) temporal scan: one thread per (b, o)
    snn_scan<<<(B_ * D2_) / 256, 256, 0, stream>>>(spk, vmem, cur);
}
